// INT8TrueMatmul_10943576670804
// MI455X (gfx1250) — 8-bit matrix operand order repaired and checked by reading, compiled, not run on hardware
//
#include <hip/hip_runtime.h>
#include <stdint.h>

typedef __attribute__((ext_vector_type(8))) int v8i;

#define LDS_ROW_STRIDE 80
#define TILE_BYTES (128 * LDS_ROW_STRIDE)

static __device__ __forceinline__ uint32_t lds_off_of(const void* p) {
  return (uint32_t)(uintptr_t)(__attribute__((address_space(3))) const void*)p;
}

static __device__ __forceinline__ void async_cp16(uint32_t lds_byte, const int8_t* g) {
  asm volatile("global_load_async_to_lds_b128 %0, %1, off"
               :: "v"(lds_byte), "v"(g)
               : "memory");
}

static __device__ __forceinline__ void wait_async0() {
#if __has_builtin(__builtin_amdgcn_s_wait_asynccnt)
  __builtin_amdgcn_s_wait_asynccnt(0);
#else
  asm volatile("s_wait_asynccnt 0x0" ::: "memory");
#endif
}

__global__ __launch_bounds__(256) void quant_rows(const float* __restrict__ in,
                                                  int8_t* __restrict__ outq,
                                                  float* __restrict__ outscale,
                                                  int K) {
  __shared__ float red[256];
  const int row = blockIdx.x;
  const int t = threadIdx.x;
  const float4* rowv = (const float4*)(in + (size_t)row * K);
  const int KV = K >> 2;

  float amax = 0.0f;
  for (int i = t; i < KV; i += 256) {
    float4 v = rowv[i];
    amax = fmaxf(amax, fabsf(v.x));
    amax = fmaxf(amax, fabsf(v.y));
    amax = fmaxf(amax, fabsf(v.z));
    amax = fmaxf(amax, fabsf(v.w));
  }
  red[t] = amax;
  __syncthreads();
  #pragma unroll
  for (int s = 128; s > 0; s >>= 1) {
    if (t < s) red[t] = fmaxf(red[t], red[t + s]);
    __syncthreads();
  }
  const float scale = fmaxf(red[0], 1e-8f) / 127.0f;
  const float inv = 1.0f / scale;
  if (t == 0) outscale[row] = scale;

  char4* outv = (char4*)(outq + (size_t)row * K);
  for (int i = t; i < KV; i += 256) {
    float4 v = rowv[i];
    char4 q;
    q.x = (int8_t)fminf(fmaxf(rintf(v.x * inv), -128.0f), 127.0f);
    q.y = (int8_t)fminf(fmaxf(rintf(v.y * inv), -128.0f), 127.0f);
    q.z = (int8_t)fminf(fmaxf(rintf(v.z * inv), -128.0f), 127.0f);
    q.w = (int8_t)fminf(fmaxf(rintf(v.w * inv), -128.0f), 127.0f);
    outv[i] = q;
  }
}

static __device__ __forceinline__ v8i load_a_frag(const int8_t* tile, int row, int g) {
  const int8_t* p = tile + row * LDS_ROW_STRIDE + g * 16;
  uint2 a01 = *(const uint2*)(p);
  uint2 a23 = *(const uint2*)(p + 8);
  uint2 a45 = *(const uint2*)(p + 32);
  uint2 a67 = *(const uint2*)(p + 40);
  return (v8i){(int)a01.x, (int)a01.y, (int)a23.x, (int)a23.y,
               (int)a45.x, (int)a45.y, (int)a67.x, (int)a67.y};
}

static __device__ __forceinline__ v8i load_b_frag(const int8_t* tile, int row, int g) {
  const int8_t* p = tile + row * LDS_ROW_STRIDE + g * 16;
  uint4 b03 = *(const uint4*)(p);
  uint4 b47 = *(const uint4*)(p + 32);
  return (v8i){(int)b03.x, (int)b03.y, (int)b03.z, (int)b03.w,
               (int)b47.x, (int)b47.y, (int)b47.z, (int)b47.w};
}

__global__ __launch_bounds__(256) void gemm_i8(const int8_t* __restrict__ xq,
                                               const int8_t* __restrict__ wq,
                                               const float* __restrict__ xs,
                                               const float* __restrict__ ws,
                                               float* __restrict__ out,
                                               int M, int N, int K) {
  __shared__ int8_t smem[4 * TILE_BYTES];

  const int t = threadIdx.x;
  const int lane = t & 31;
  const int wave = t >> 5;
  const int g = lane >> 4;
  const int ml = lane & 15;
  const int wm = wave >> 1;
  const int wn = wave & 1;

  const int m0 = blockIdx.y * 128;
  const int n0 = blockIdx.x * 128;

  const int8_t* gA = xq + (size_t)m0 * K;
  const int8_t* gB = wq + (size_t)n0 * K;

  const int r0 = t >> 2, c0 = (t & 3) << 4;
  const int r1 = (t + 256) >> 2, c1 = ((t + 256) & 3) << 4;
  const uint32_t sbase = lds_off_of(smem);

  v8i acc[2][4];
  #pragma unroll
  for (int i = 0; i < 2; ++i)
    #pragma unroll
    for (int j = 0; j < 4; ++j) acc[i][j] = (v8i){0, 0, 0, 0, 0, 0, 0, 0};

  {
    const uint32_t aL = sbase;
    const uint32_t bL = sbase + TILE_BYTES;
    async_cp16(aL + r0 * LDS_ROW_STRIDE + c0, gA + (size_t)r0 * K + c0);
    async_cp16(aL + r1 * LDS_ROW_STRIDE + c1, gA + (size_t)r1 * K + c1);
    async_cp16(bL + r0 * LDS_ROW_STRIDE + c0, gB + (size_t)r0 * K + c0);
    async_cp16(bL + r1 * LDS_ROW_STRIDE + c1, gB + (size_t)r1 * K + c1);
  }

  for (int k = 0; k < K; k += 64) {
    wait_async0();
    __syncthreads();

    const int cur = (k >> 6) & 1;

    if (k + 64 < K) {
      const uint32_t aL = sbase + (uint32_t)(cur ^ 1) * (2 * TILE_BYTES);
      const uint32_t bL = aL + TILE_BYTES;
      const int8_t* gAk = gA + (k + 64);
      const int8_t* gBk = gB + (k + 64);
      async_cp16(aL + r0 * LDS_ROW_STRIDE + c0, gAk + (size_t)r0 * K + c0);
      async_cp16(aL + r1 * LDS_ROW_STRIDE + c1, gAk + (size_t)r1 * K + c1);
      async_cp16(bL + r0 * LDS_ROW_STRIDE + c0, gBk + (size_t)r0 * K + c0);
      async_cp16(bL + r1 * LDS_ROW_STRIDE + c1, gBk + (size_t)r1 * K + c1);
    }

    const int8_t* At = smem + cur * (2 * TILE_BYTES);
    const int8_t* Bt = At + TILE_BYTES;

    v8i a0 = load_a_frag(At, wm * 32 + ml, g);
    v8i a1 = load_a_frag(At, wm * 32 + 16 + ml, g);

    #pragma unroll
    for (int j = 0; j < 4; ++j) {
      v8i b = load_b_frag(Bt, wn * 64 + j * 16 + ml, g);
      acc[0][j] = __builtin_amdgcn_wmma_i32_16x16x64_iu8(true, a0, true, b,
                                                         acc[0][j], false, false);
      acc[1][j] = __builtin_amdgcn_wmma_i32_16x16x64_iu8(true, a1, true, b,
                                                         acc[1][j], false, false);
    }
  }

  float xsr[2][8];
  #pragma unroll
  for (int i = 0; i < 2; ++i)
    #pragma unroll
    for (int r = 0; r < 8; ++r)
      xsr[i][r] = xs[m0 + wm * 32 + i * 16 + g * 8 + r];

  #pragma unroll
  for (int i = 0; i < 2; ++i) {
    #pragma unroll
    for (int j = 0; j < 4; ++j) {
      const int n = n0 + wn * 64 + j * 16 + ml;
      const float wsc = ws[n];
      float* po = out + n;
      #pragma unroll
      for (int r = 0; r < 8; ++r) {
        const int m = m0 + wm * 32 + i * 16 + g * 8 + r;
        po[(size_t)m * N] = (float)acc[i][j][r] * xsr[i][r] * wsc;
      }
    }
  }
}

extern "C" void kernel_launch(void* const* d_in, const int* in_sizes, int n_in,
                              void* d_out, int out_size, void* d_ws, size_t ws_size,
                              hipStream_t stream) {
  const float* x = (const float*)d_in[0];
  const float* w = (const float*)d_in[1];
  float* out = (float*)d_out;

  const int K = 4096;
  const int M = in_sizes[0] / K;
  const int N = in_sizes[1] / K;

  uint8_t* ws8 = (uint8_t*)d_ws;
  float* x_scale = (float*)ws8;
  float* w_scale = (float*)(ws8 + (size_t)M * 4);
  int8_t* xq = (int8_t*)(ws8 + (size_t)(M + N) * 4);
  int8_t* wq = xq + (size_t)M * K;

  quant_rows<<<N, 256, 0, stream>>>(w, wq, w_scale, K);
  quant_rows<<<M, 256, 0, stream>>>(x, xq, x_scale, K);

  dim3 grid(N / 128, M / 128);
  gemm_i8<<<grid, 256, 0, stream>>>(xq, wq, x_scale, w_scale, out, M, N, K);
}
